// TopologicalPredictor_27092653703530
// MI455X (gfx1250) — matrix operand order repaired and checked by reading, not compiled, not run on hardware
//
#include <hip/hip_runtime.h>

// ---------------------------------------------------------------------------
// TopologicalPredictor (GNN message passing) for MI455X / gfx1250, wave32.
//  * node_mlp(h[src]) == node_mlp(h)[src]  -> per-node GEMM, not per-edge.
//  * e embedded once, stored f16, streamed 3x.
//  * All MLP matmuls via v_wmma_f32_16x16x32_f16 (f32 accum).
//  * LDS weight/staging rows padded (72/136 halves) to kill bank conflicts.
//  * Edge kernel: 32-edge macro-tiles, two A-tiles share every B fragment.
// ---------------------------------------------------------------------------

#define NN 50000
#define NE 1600000
#define NG 64
#define HD 64

#define KP64  72    // padded LDS stride for K=64  rows (144B: banks 36n%64, conflict-free)
#define KP128 136   // padded LDS stride for K=128 rows (272B: banks 68n%64, conflict-free)

typedef __attribute__((ext_vector_type(16))) _Float16 v16h;
typedef __attribute__((ext_vector_type(8)))  _Float16 v8h;
typedef __attribute__((ext_vector_type(8)))  float    v8f;

union AV { v16h v; v8h h[2]; };

// A fragment (16x32, 16-bit): lane l holds row m=l&15,
// K = kb..kb+7 and kb+16..kb+23 where kb = kt*32 + (l>=16 ? 8 : 0).
// Requires 16B alignment of base + m*stride (stride in halves).
__device__ __forceinline__ v16h loadA(const _Float16* base, int stride, int kt, int lane) {
  const int m  = lane & 15;
  const int kb = kt * 32 + ((lane & 16) ? 8 : 0);
  AV a;
  a.h[0] = *(const v8h*)(base + m * stride + kb);
  a.h[1] = *(const v8h*)(base + m * stride + kb + 16);
  return a.v;
}

// B fragment (32x16, 16-bit) from LDS-transposed weights Wt[N][Kp] (padded):
// lane l holds column n = nt*16 + (l&15), same K order as loadA:
// K = kb..kb+7 and kb+16..kb+23 where kb = kt*32 + (l>=16 ? 8 : 0).
__device__ __forceinline__ v16h loadB(const _Float16* wt, int Kp, int nt, int kt, int lane) {
  const int n  = nt * 16 + (lane & 15);
  const int kb = kt * 32 + ((lane & 16) ? 8 : 0);
  AV a;
  a.h[0] = *(const v8h*)(wt + n * Kp + kb);
  a.h[1] = *(const v8h*)(wt + n * Kp + kb + 16);
  return a.v;
}

#define WMMA_F16(a, b, c) \
  __builtin_amdgcn_wmma_f32_16x16x32_f16(false, (a), false, (b), (short)0, (c), false, false)

// ---------------------------------------------------------------------------
// Weight convert + transpose (global, unpadded): Wt[n*K + k] = (f16) W[k*N + n]
// ---------------------------------------------------------------------------
__global__ void cvtT_kernel(const float* __restrict__ W, _Float16* __restrict__ Wt,
                            int K, int N) {
  int i = blockIdx.x * blockDim.x + threadIdx.x;
  if (i < K * N) {
    int n = i / K, k = i % K;
    Wt[i] = (_Float16)W[k * N + n];
  }
}

__global__ void zero_f32_kernel(float* __restrict__ p, int n) {
  int i = blockIdx.x * blockDim.x + threadIdx.x;
  if (i < n) p[i] = 0.0f;
}

__global__ void embed_nodes_kernel(const float* __restrict__ x,
                                   const float* __restrict__ W,
                                   const float* __restrict__ b,
                                   float* __restrict__ h,
                                   _Float16* __restrict__ hcat) {
  int i = blockIdx.x * blockDim.x + threadIdx.x;
  if (i >= NN * HD) return;
  int n = i / HD, c = i % HD;
  float acc = b[c];
  #pragma unroll
  for (int k = 0; k < 12; ++k) acc += x[n * 12 + k] * W[k * HD + c];
  h[i] = acc;
  hcat[(size_t)n * 128 + c] = (_Float16)acc;
}

__global__ void embed_edges_kernel(const float* __restrict__ ea,
                                   const float* __restrict__ W,
                                   const float* __restrict__ b,
                                   _Float16* __restrict__ e16) {
  int i = blockIdx.x * blockDim.x + threadIdx.x;
  if (i >= NE * HD) return;
  int e = i / HD, c = i % HD;
  float acc = b[c];
  #pragma unroll
  for (int k = 0; k < 20; ++k) acc += ea[(size_t)e * 20 + k] * W[k * HD + c];
  e16[i] = (_Float16)acc;
}

__global__ void aggr_to_hcat_kernel(const float* __restrict__ aggr,
                                    _Float16* __restrict__ hcat) {
  int i = blockIdx.x * blockDim.x + threadIdx.x;
  if (i >= NN * HD) return;
  int n = i / HD, c = i % HD;
  hcat[(size_t)n * 128 + 64 + c] = (_Float16)aggr[i];
}

// Cooperative copy of Wt[N][K] (global) into padded LDS Wt[N][Kp].
__device__ __forceinline__ void ldsCopyW(_Float16* dst, const _Float16* src,
                                         int K, int Kp, int N, int tid, int nthr) {
  for (int i = tid; i < N * K; i += nthr) {
    int n = i / K, k = i - n * K;
    dst[n * Kp + k] = src[i];
  }
}

// ---------------------------------------------------------------------------
// node_mlp: out = relu(h @ W1 + b1) @ W2 + b2, rows = NN, K=64->128->64
// ---------------------------------------------------------------------------
__global__ void __launch_bounds__(128, 1)
node_mlp_kernel(const _Float16* __restrict__ hcat,
                const _Float16* __restrict__ W1t, const float* __restrict__ b1,
                const _Float16* __restrict__ W2t, const float* __restrict__ b2,
                float* __restrict__ outv, int ntiles) {
  __shared__ __align__(32) _Float16 sW1[128 * KP64];     // [128 out][64 in]
  __shared__ __align__(32) _Float16 sW2[64 * KP128];     // [64 out][128 in]
  __shared__ __align__(32) _Float16 stg[4][16 * KP128];
  ldsCopyW(sW1, W1t, 64, KP64, 128, threadIdx.x, 128);
  ldsCopyW(sW2, W2t, 128, KP128, 64, threadIdx.x, 128);
  __syncthreads();
  const int lane = threadIdx.x & 31;
  const int wv   = threadIdx.x >> 5;
  _Float16* st = stg[wv];
  const int nl = lane & 15;
  const int mr = (lane & 16) ? 8 : 0;

  for (int tile = blockIdx.x * 4 + wv; tile < ntiles; tile += gridDim.x * 4) {
    const int rb = tile * 16;
    v16h a0 = loadA(hcat + (size_t)rb * 128, 128, 0, lane);
    v16h a1 = loadA(hcat + (size_t)rb * 128, 128, 1, lane);
    #pragma unroll
    for (int nt = 0; nt < 8; ++nt) {
      v8f c = {};
      c = WMMA_F16(a0, loadB(sW1, KP64, nt, 0, lane), c);
      c = WMMA_F16(a1, loadB(sW1, KP64, nt, 1, lane), c);
      const int n = nt * 16 + nl;
      const float bb = b1[n];
      #pragma unroll
      for (int v = 0; v < 8; ++v) {
        float xv = c[v] + bb;
        st[(mr + v) * KP128 + n] = (_Float16)(xv > 0.0f ? xv : 0.0f);
      }
    }
    v8f z = {};
    v8f d[4] = {z, z, z, z};
    #pragma unroll
    for (int kt = 0; kt < 4; ++kt) {
      v16h a = loadA(st, KP128, kt, lane);
      #pragma unroll
      for (int nt = 0; nt < 4; ++nt)
        d[nt] = WMMA_F16(a, loadB(sW2, KP128, nt, kt, lane), d[nt]);
    }
    #pragma unroll
    for (int v = 0; v < 8; ++v) {
      const int row = rb + mr + v;
      #pragma unroll
      for (int nt = 0; nt < 4; ++nt) {
        const int n = nt * 16 + nl;
        outv[(size_t)row * HD + n] = d[nt][v] + b2[n];
      }
    }
  }
}

// ---------------------------------------------------------------------------
// edge kernel, 32-edge macro-tiles (two 16-row A-tiles share every B fragment):
//   edge_out = relu(e @ W1 + b1) @ W2 + b2
//   msg = edge_out * node_out[src]; aggr[dst] += msg (hardware f32 atomics)
// 3 waves / block: LDS = 18.4K + 17.4K + 3*8.7K + idx ~= 62.7KB
// ---------------------------------------------------------------------------
__global__ void __launch_bounds__(96, 1)
edge_layer_kernel(const _Float16* __restrict__ e16,
                  const _Float16* __restrict__ W1t, const float* __restrict__ b1,
                  const _Float16* __restrict__ W2t, const float* __restrict__ b2,
                  const float* __restrict__ node_out,
                  const int* __restrict__ ei,
                  float* __restrict__ aggr, int ntiles) {
  __shared__ __align__(32) _Float16 sW1[128 * KP64];
  __shared__ __align__(32) _Float16 sW2[64 * KP128];
  __shared__ __align__(32) _Float16 stg[3][32 * KP128];
  __shared__ int sidx[3][32], didx[3][32];
  ldsCopyW(sW1, W1t, 64, KP64, 128, threadIdx.x, 96);
  ldsCopyW(sW2, W2t, 128, KP128, 64, threadIdx.x, 96);
  __syncthreads();
  const int lane = threadIdx.x & 31;
  const int wv   = threadIdx.x >> 5;
  _Float16* st = stg[wv];
  const int nl = lane & 15;
  const int mr = (lane & 16) ? 8 : 0;

  for (int tile = blockIdx.x * 3 + wv; tile < ntiles; tile += gridDim.x * 3) {
    const int eb = tile * 32;
    // stage scaled src/dst indices (one per lane; intra-wave LDS is in-order)
    sidx[wv][lane] = ei[eb + lane] * HD;
    didx[wv][lane] = ei[NE + eb + lane] * HD;
    // prefetch next macro-tile of e16
    int nxt = tile + gridDim.x * 3;
    if (nxt < ntiles) __builtin_prefetch(e16 + (size_t)nxt * 32 * HD, 0, 1);

    const _Float16* erow = e16 + (size_t)eb * HD;
    v16h a0 = loadA(erow, HD, 0, lane);
    v16h a1 = loadA(erow, HD, 1, lane);
    v16h b0 = loadA(erow + 16 * HD, HD, 0, lane);
    v16h b1f = loadA(erow + 16 * HD, HD, 1, lane);

    // GEMM1 + ReLU -> stage (rows 0..15 = tile A, rows 16..31 = tile B)
    #pragma unroll
    for (int nt = 0; nt < 8; ++nt) {
      v16h w0 = loadB(sW1, KP64, nt, 0, lane);
      v16h w1 = loadB(sW1, KP64, nt, 1, lane);
      v8f cA = {}, cB = {};
      cA = WMMA_F16(a0, w0, cA);
      cB = WMMA_F16(b0, w0, cB);
      cA = WMMA_F16(a1, w1, cA);
      cB = WMMA_F16(b1f, w1, cB);
      const int n = nt * 16 + nl;
      const float bb = b1[n];
      #pragma unroll
      for (int v = 0; v < 8; ++v) {
        float xA = cA[v] + bb;
        float xB = cB[v] + bb;
        st[(mr + v) * KP128 + n]        = (_Float16)(xA > 0.0f ? xA : 0.0f);
        st[(16 + mr + v) * KP128 + n]   = (_Float16)(xB > 0.0f ? xB : 0.0f);
      }
    }
    // GEMM2: (2 x 16x128) @ 128x64, B fragments shared across both tiles
    v8f z = {};
    v8f dA[4] = {z, z, z, z};
    v8f dB[4] = {z, z, z, z};
    #pragma unroll
    for (int kt = 0; kt < 4; ++kt) {
      v16h aa = loadA(st, KP128, kt, lane);
      v16h ab = loadA(st + 16 * KP128, KP128, kt, lane);
      #pragma unroll
      for (int nt = 0; nt < 4; ++nt) {
        v16h w = loadB(sW2, KP128, nt, kt, lane);
        dA[nt] = WMMA_F16(aa, w, dA[nt]);
        dB[nt] = WMMA_F16(ab, w, dB[nt]);
      }
    }
    // epilogue: multiply by gathered node_out[src], scatter-add to aggr[dst]
    #pragma unroll
    for (int v = 0; v < 8; ++v) {
      const int mA = mr + v;
      const int soA  = sidx[wv][mA],      dofA = didx[wv][mA];
      const int soB  = sidx[wv][16 + mA], dofB = didx[wv][16 + mA];
      #pragma unroll
      for (int nt = 0; nt < 4; ++nt) {
        const int n = nt * 16 + nl;
        float vA = (dA[nt][v] + b2[n]) * node_out[soA + n];
        float vB = (dB[nt][v] + b2[n]) * node_out[soB + n];
        __hip_atomic_fetch_add(&aggr[dofA + n], vA, __ATOMIC_RELAXED,
                               __HIP_MEMORY_SCOPE_AGENT);
        __hip_atomic_fetch_add(&aggr[dofB + n], vB, __ATOMIC_RELAXED,
                               __HIP_MEMORY_SCOPE_AGENT);
      }
    }
  }
}

// ---------------------------------------------------------------------------
// combine: h = relu(concat(h,aggr) @ W1 + b1) @ W2 + b2  (K=128->64->64)
// ---------------------------------------------------------------------------
__global__ void __launch_bounds__(128, 1)
combine_kernel(_Float16* __restrict__ hcat,
               const _Float16* __restrict__ W1t, const float* __restrict__ b1, // [64][128]
               const _Float16* __restrict__ W2t, const float* __restrict__ b2, // [64][64]
               float* __restrict__ h, int ntiles) {
  __shared__ __align__(32) _Float16 sW1[64 * KP128];
  __shared__ __align__(32) _Float16 sW2[64 * KP64];
  __shared__ __align__(32) _Float16 stg[4][16 * KP64];
  ldsCopyW(sW1, W1t, 128, KP128, 64, threadIdx.x, 128);
  ldsCopyW(sW2, W2t, 64, KP64, 64, threadIdx.x, 128);
  __syncthreads();
  const int lane = threadIdx.x & 31;
  const int wv   = threadIdx.x >> 5;
  _Float16* st = stg[wv];
  const int nl = lane & 15;
  const int mr = (lane & 16) ? 8 : 0;

  for (int tile = blockIdx.x * 4 + wv; tile < ntiles; tile += gridDim.x * 4) {
    const int rb = tile * 16;
    v16h a[4];
    #pragma unroll
    for (int kt = 0; kt < 4; ++kt) a[kt] = loadA(hcat + (size_t)rb * 128, 128, kt, lane);
    #pragma unroll
    for (int nt = 0; nt < 4; ++nt) {
      v8f c = {};
      #pragma unroll
      for (int kt = 0; kt < 4; ++kt)
        c = WMMA_F16(a[kt], loadB(sW1, KP128, nt, kt, lane), c);
      const int n = nt * 16 + nl;
      const float bb = b1[n];
      #pragma unroll
      for (int v = 0; v < 8; ++v) {
        float xv = c[v] + bb;
        st[(mr + v) * KP64 + n] = (_Float16)(xv > 0.0f ? xv : 0.0f);
      }
    }
    v8f z = {};
    v8f d[4] = {z, z, z, z};
    #pragma unroll
    for (int kt = 0; kt < 2; ++kt) {
      v16h aa = loadA(st, KP64, kt, lane);
      #pragma unroll
      for (int nt = 0; nt < 4; ++nt)
        d[nt] = WMMA_F16(aa, loadB(sW2, KP64, nt, kt, lane), d[nt]);
    }
    #pragma unroll
    for (int v = 0; v < 8; ++v) {
      const int row = rb + mr + v;
      #pragma unroll
      for (int nt = 0; nt < 4; ++nt) {
        const int n = nt * 16 + nl;
        float val = d[nt][v] + b2[n];
        h[(size_t)row * HD + n] = val;
        hcat[(size_t)row * 128 + n] = (_Float16)val;
      }
    }
  }
}

// ---------------------------------------------------------------------------
// Pooling + head
// ---------------------------------------------------------------------------
__global__ void pool_accum_kernel(const float* __restrict__ h,
                                  const int* __restrict__ batch,
                                  float* __restrict__ pooled,
                                  float* __restrict__ counts) {
  int i = blockIdx.x * blockDim.x + threadIdx.x;
  if (i >= NN * HD) return;
  int n = i / HD, c = i % HD;
  int g = batch[n];
  __hip_atomic_fetch_add(&pooled[g * HD + c], h[i], __ATOMIC_RELAXED,
                         __HIP_MEMORY_SCOPE_AGENT);
  if (c == 0)
    __hip_atomic_fetch_add(&counts[g], 1.0f, __ATOMIC_RELAXED,
                           __HIP_MEMORY_SCOPE_AGENT);
}

__global__ void head_kernel(const float* __restrict__ pooled,
                            const float* __restrict__ counts,
                            const float* __restrict__ rW1, const float* __restrict__ rb1,
                            const float* __restrict__ rW2, const float* __restrict__ rb2,
                            const float* __restrict__ oW1, const float* __restrict__ ob1,
                            const float* __restrict__ oW2, const float* __restrict__ ob2,
                            float* __restrict__ out) {
  int g = threadIdx.x;
  if (g >= NG) return;
  float cnt = counts[g];
  if (cnt < 1.0f) cnt = 1.0f;
  float mean[64];
  for (int c = 0; c < 64; ++c) mean[c] = pooled[g * 64 + c] / cnt;
  float t[128];
  for (int j = 0; j < 128; ++j) {
    float acc = rb1[j];
    for (int c = 0; c < 64; ++c) acc += mean[c] * rW1[c * 128 + j];
    t[j] = acc > 0.0f ? acc : 0.0f;
  }
  float r[64];
  for (int j = 0; j < 64; ++j) {
    float acc = rb2[j];
    for (int k = 0; k < 128; ++k) acc += t[k] * rW2[k * 64 + j];
    r[j] = acc;
  }
  float o[64];
  for (int j = 0; j < 64; ++j) {
    float acc = ob1[j];
    for (int c = 0; c < 64; ++c) acc += r[c] * oW1[c * 64 + j];
    o[j] = acc > 0.0f ? acc : 0.0f;
  }
  for (int q = 0; q < 7; ++q) {
    float acc = ob2[q];
    for (int c = 0; c < 64; ++c) acc += o[c] * oW2[c * 7 + q];
    out[g * 7 + q] = acc;
  }
}

// ---------------------------------------------------------------------------
// Host-side orchestration
// ---------------------------------------------------------------------------
extern "C" void kernel_launch(void* const* d_in, const int* in_sizes, int n_in,
                              void* d_out, int out_size, void* d_ws, size_t ws_size,
                              hipStream_t stream) {
  // Input order: x, edge_attr, edge_index, batch, then params pytree leaves
  // (dicts flattened by sorted key): edge_emb, layers[0..2]{combine,edge_mlp,
  // node_mlp}, node_emb, output, readout; each Linear = (W, b).
  const float* x     = (const float*)d_in[0];
  const float* ea    = (const float*)d_in[1];
  const int*   ei    = (const int*)d_in[2];
  const int*   batch = (const int*)d_in[3];
  const float* eeW   = (const float*)d_in[4];   // [20,64]
  const float* eeB   = (const float*)d_in[5];
  const float* neW   = (const float*)d_in[42];  // [12,64]
  const float* neB   = (const float*)d_in[43];
  const float* oW1 = (const float*)d_in[44], *ob1 = (const float*)d_in[45];
  const float* oW2 = (const float*)d_in[46], *ob2 = (const float*)d_in[47];
  const float* rW1 = (const float*)d_in[48], *rb1 = (const float*)d_in[49];
  const float* rW2 = (const float*)d_in[50], *rb2 = (const float*)d_in[51];

  // Workspace carve-up
  char* ws = (char*)d_ws;
  size_t off = 0;
  float* h       = (float*)(ws + off);    off += (size_t)NN * HD * 4;   // 12.8 MB
  float* node_o  = (float*)(ws + off);    off += (size_t)NN * HD * 4;   // 12.8 MB
  float* aggr    = (float*)(ws + off);    off += (size_t)NN * HD * 4;   // 12.8 MB
  _Float16* hcat = (_Float16*)(ws + off); off += (size_t)NN * 128 * 2;  // 12.8 MB
  float* pooled  = (float*)(ws + off);    off += (size_t)NG * HD * 4;
  float* counts  = (float*)(ws + off);    off += 256;
  _Float16* wbuf = (_Float16*)(ws + off); off += (size_t)3 * 45056 * 2;
  off = (off + 255) & ~(size_t)255;
  _Float16* e16  = (_Float16*)(ws + off); off += (size_t)NE * HD * 2;   // 204.8 MB

  const int T = 256;

  // 1) Weight prep: convert+transpose per-layer matmul weights to f16 [N][K].
  for (int l = 0; l < 3; ++l) {
    const int b = 6 + 12 * l; // combine.W1,b1,W2,b2, edge.W1,b1,W2,b2, node.W1,b1,W2,b2
    _Float16* wl = wbuf + (size_t)l * 45056;
    cvtT_kernel<<<(64 * 128 + T - 1) / T, T, 0, stream>>>((const float*)d_in[b + 8],  wl + 0,     64, 128);  // node W1
    cvtT_kernel<<<(128 * 64 + T - 1) / T, T, 0, stream>>>((const float*)d_in[b + 10], wl + 8192, 128, 64);   // node W2
    cvtT_kernel<<<(64 * 128 + T - 1) / T, T, 0, stream>>>((const float*)d_in[b + 4],  wl + 16384, 64, 128);  // edge W1
    cvtT_kernel<<<(128 * 64 + T - 1) / T, T, 0, stream>>>((const float*)d_in[b + 6],  wl + 24576, 128, 64);  // edge W2
    cvtT_kernel<<<(128 * 64 + T - 1) / T, T, 0, stream>>>((const float*)d_in[b + 0],  wl + 32768, 128, 64);  // comb W1
    cvtT_kernel<<<(64 * 64 + T - 1) / T, T, 0, stream>>>((const float*)d_in[b + 2],   wl + 40960, 64, 64);   // comb W2
  }

  // 2) Embeddings
  embed_nodes_kernel<<<(NN * HD + T - 1) / T, T, 0, stream>>>(x, neW, neB, h, hcat);
  embed_edges_kernel<<<((size_t)NE * HD + T - 1) / T, T, 0, stream>>>(ea, eeW, eeB, e16);

  // 3) Message-passing layers
  const int node_tiles = NN / 16;   // 3125
  const int edge_tiles = NE / 32;   // 50000 macro-tiles of 32 edges
  for (int l = 0; l < 3; ++l) {
    const int b = 6 + 12 * l;
    _Float16* wl = wbuf + (size_t)l * 45056;
    zero_f32_kernel<<<(NN * HD + T - 1) / T, T, 0, stream>>>(aggr, NN * HD);
    node_mlp_kernel<<<782, 128, 0, stream>>>(
        hcat, wl + 0, (const float*)d_in[b + 9], wl + 8192, (const float*)d_in[b + 11],
        node_o, node_tiles);
    edge_layer_kernel<<<1024, 96, 0, stream>>>(
        e16, wl + 16384, (const float*)d_in[b + 5], wl + 24576, (const float*)d_in[b + 7],
        node_o, ei, aggr, edge_tiles);
    aggr_to_hcat_kernel<<<(NN * HD + T - 1) / T, T, 0, stream>>>(aggr, hcat);
    combine_kernel<<<782, 128, 0, stream>>>(
        hcat, wl + 32768, (const float*)d_in[b + 1], wl + 40960, (const float*)d_in[b + 3],
        h, node_tiles);
  }

  // 4) Global mean pool + readout + output head
  zero_f32_kernel<<<(NG * HD + 64 + T - 1) / T, T, 0, stream>>>(pooled, NG * HD + 64);
  pool_accum_kernel<<<(NN * HD + T - 1) / T, T, 0, stream>>>(h, batch, pooled, counts);
  head_kernel<<<1, 64, 0, stream>>>(pooled, counts, rW1, rb1, rW2, rb2,
                                    oW1, ob1, oW2, ob2, (float*)d_out);
}
